// GCN_83657372991833
// MI455X (gfx1250) — matrix operand order repaired and checked by reading, not compiled, not run on hardware
//
#include <hip/hip_runtime.h>
#include <hip/hip_bf16.h>
#include <math.h>

// ---------------------------------------------------------------------------
// GCN forward for MI455X (gfx1250, wave32).
// Matmuls run on v_wmma_f32_16x16x32_bf16 (bf16 inputs, fp32 accumulate).
// Edge aggregation is streaming gather + global_atomic_add_f32 scatter.
// ---------------------------------------------------------------------------

typedef __attribute__((ext_vector_type(16))) __bf16 v16bf;
typedef __attribute__((ext_vector_type(16))) short  v16s;
typedef __attribute__((ext_vector_type(8)))  float  v8f;

__device__ __forceinline__ unsigned short bf16rne(float f) {
  unsigned u = __builtin_bit_cast(unsigned, f);
  u += 0x7FFFu + ((u >> 16) & 1u);          // round-to-nearest-even
  return (unsigned short)(u >> 16);
}

// ---------------------------------------------------------------------------
// Degree / symmetric normalization  (deg starts at 1.0 for the self-loop)
// ---------------------------------------------------------------------------
__global__ void gcn_deg_init(float* __restrict__ deg, int n) {
  int i = blockIdx.x * blockDim.x + threadIdx.x;
  if (i < n) deg[i] = 1.0f;
}

__global__ void gcn_deg_count(const int* __restrict__ ei, float* __restrict__ deg, int E) {
  int e = blockIdx.x * blockDim.x + threadIdx.x;
  if (e < E) atomicAdd(&deg[ei[E + e]], 1.0f);   // dst = edge_index[1][e]
}

__global__ void gcn_deg_rsqrt(float* __restrict__ deg, int n) {
  int i = blockIdx.x * blockDim.x + threadIdx.x;
  if (i < n) {
    float d = deg[i];
    deg[i] = d > 0.0f ? rsqrtf(d) : 0.0f;        // becomes dinv
  }
}

// ---------------------------------------------------------------------------
// Pack W [FIN, FOUT_REAL] (f32, row-major) into bf16 WMMA B-fragments:
// layout[((ct*KT + kt)*32 + lane)*16 + j],  k = kt*32 + 16*(j/8) + 8*(lane/16) + (j%8),
// col = ct*16 + (lane%16).  Out-of-range columns packed as 0.
// ---------------------------------------------------------------------------
__global__ void gcn_pack_w(const float* __restrict__ W, unsigned short* __restrict__ out,
                           int FIN, int FOUT_REAL, int KT, int CT) {
  int tid = blockIdx.x * blockDim.x + threadIdx.x;
  int total = CT * KT * 512;
  if (tid >= total) return;
  int j    = tid & 15;
  int lane = (tid >> 4) & 31;
  int tile = tid >> 9;
  int kt = tile % KT;
  int ct = tile / KT;
  int l = lane & 15, h = lane >> 4;
  int k   = kt * 32 + 16 * (j >> 3) + 8 * h + (j & 7);
  int col = ct * 16 + l;
  float v = 0.0f;
  if (k < FIN && col < FOUT_REAL) v = W[(size_t)k * FOUT_REAL + col];
  out[tid] = bf16rne(v);
}

// ---------------------------------------------------------------------------
// WMMA GEMM:  Out[N, FOUT_REAL] = (opt relu(A)) @ W  (+ opt bias)
// One wave owns a 16-row strip; A fragments for the whole K dim live in VGPRs.
// 4 waves per block.  nRowTiles = N/16 (N is a multiple of 16).
// ---------------------------------------------------------------------------
template<int FIN, int FOUT_PAD, int FOUT_REAL, bool ADD_BIAS, bool RELU_IN>
__global__ __launch_bounds__(128)
void gcn_matmul_wmma(const float* __restrict__ A,
                     const unsigned short* __restrict__ Bpack,
                     const float* __restrict__ bias,
                     float* __restrict__ Out,
                     int nRowTiles) {
  constexpr int KT = FIN / 32;
  constexpr int CT = FOUT_PAD / 16;
  int wave = threadIdx.x >> 5;
  int lane = threadIdx.x & 31;
  int rt = blockIdx.x * 4 + wave;
  if (rt >= nRowTiles) return;
  int l = lane & 15;
  int h = lane >> 4;

  const float* arow = A + (size_t)(rt * 16 + l) * FIN;

  // A fragment: lane l holds row l; element j -> k = kt*32 + 16*(j/8) + 8*h + (j%8)
  v16bf afrag[KT];
#pragma unroll
  for (int kt = 0; kt < KT; ++kt) {
    int k0 = kt * 32 + 8 * h;
    float4 a0 = *(const float4*)(arow + k0);
    float4 a1 = *(const float4*)(arow + k0 + 4);
    float4 a2 = *(const float4*)(arow + k0 + 16);
    float4 a3 = *(const float4*)(arow + k0 + 20);
    float vv[16] = {a0.x, a0.y, a0.z, a0.w, a1.x, a1.y, a1.z, a1.w,
                    a2.x, a2.y, a2.z, a2.w, a3.x, a3.y, a3.z, a3.w};
    v16s s;
#pragma unroll
    for (int j = 0; j < 16; ++j) {
      float f = vv[j];
      if (RELU_IN) f = f > 0.0f ? f : 0.0f;
      s[j] = (short)bf16rne(f);
    }
    afrag[kt] = __builtin_bit_cast(v16bf, s);
  }

  const v16s* bp = (const v16s*)Bpack;
#pragma unroll
  for (int ct = 0; ct < CT; ++ct) {
    int col = ct * 16 + l;
    float binit = 0.0f;
    if (ADD_BIAS) binit = (col < FOUT_REAL) ? bias[col] : 0.0f;
    v8f acc;
#pragma unroll
    for (int r = 0; r < 8; ++r) acc[r] = binit;

#pragma unroll
    for (int kt = 0; kt < KT; ++kt) {
      v16bf bfrag = __builtin_bit_cast(v16bf, bp[(ct * KT + kt) * 32 + lane]);
      acc = __builtin_amdgcn_wmma_f32_16x16x32_bf16(
          false, afrag[kt], false, bfrag, (short)0, acc, false, false);
    }

    if (col < FOUT_REAL) {
#pragma unroll
      for (int r = 0; r < 8; ++r) {
        int m = r + 8 * h;                        // C/D layout: lanes 16-31 hold M=8..15
        Out[(size_t)(rt * 16 + m) * FOUT_REAL + col] = acc[r];
      }
    }
  }
}

// ---------------------------------------------------------------------------
// Aggregation init: Out[i] = H[i] * dinv[i]^2 + bias   (self-loop + bias)
// FOUT = 128 fixed; float4 vectorized: 32 float4 per node.
// ---------------------------------------------------------------------------
__global__ void gcn_agg_init(const float* __restrict__ H, const float* __restrict__ dinv,
                             const float* __restrict__ bias, float* __restrict__ Out, int n) {
  int tid = blockIdx.x * blockDim.x + threadIdx.x;
  if (tid >= n * 32) return;
  int i  = tid >> 5;
  int f4 = tid & 31;
  float w = dinv[i];
  w = w * w;
  float4 hv = ((const float4*)(H + (size_t)i * 128))[f4];
  float4 bv = ((const float4*)bias)[f4];
  float4 o;
  o.x = hv.x * w + bv.x;
  o.y = hv.y * w + bv.y;
  o.z = hv.z * w + bv.z;
  o.w = hv.w * w + bv.w;
  ((float4*)(Out + (size_t)i * 128))[f4] = o;
}

// ---------------------------------------------------------------------------
// Edge scatter: one wave per edge, 32 lanes x float4 = 128 features.
// Out[dst] += H[src] * dinv[src]*dinv[dst]   via global_atomic_add_f32.
// ---------------------------------------------------------------------------
__global__ __launch_bounds__(256)
void gcn_agg_edges(const float* __restrict__ H, const int* __restrict__ ei,
                   const float* __restrict__ dinv, float* __restrict__ Out, int E) {
  int gw   = blockIdx.x * 8 + (threadIdx.x >> 5);
  int lane = threadIdx.x & 31;
  if (gw >= E) return;
  int s = ei[gw];
  int d = ei[E + gw];
  float w = dinv[s] * dinv[d];
  float4 hv = ((const float4*)(H + (size_t)s * 128))[lane];
  float* o = Out + (size_t)d * 128 + lane * 4;
  atomicAdd(o + 0, hv.x * w);
  atomicAdd(o + 1, hv.y * w);
  atomicAdd(o + 2, hv.z * w);
  atomicAdd(o + 3, hv.w * w);
}

// ---------------------------------------------------------------------------
// Row-wise log_softmax over 40 classes.
// ---------------------------------------------------------------------------
__global__ void gcn_log_softmax(const float* __restrict__ L, float* __restrict__ O, int n) {
  int i = blockIdx.x * blockDim.x + threadIdx.x;
  if (i >= n) return;
  const float* p = L + (size_t)i * 40;
  float m = p[0];
#pragma unroll
  for (int j = 1; j < 40; ++j) m = fmaxf(m, p[j]);
  float s = 0.0f;
#pragma unroll
  for (int j = 0; j < 40; ++j) s += expf(p[j] - m);
  float ls = logf(s);
  float* o = O + (size_t)i * 40;
#pragma unroll
  for (int j = 0; j < 40; ++j) o[j] = p[j] - m - ls;
}

// ---------------------------------------------------------------------------
// Host-side launch
// ---------------------------------------------------------------------------
extern "C" void kernel_launch(void* const* d_in, const int* in_sizes, int n_in,
                              void* d_out, int out_size, void* d_ws, size_t ws_size,
                              hipStream_t stream) {
  const float* x     = (const float*)d_in[0];
  const int*   ei    = (const int*)  d_in[1];
  const float* Wpre  = (const float*)d_in[2];
  const float* bpre  = (const float*)d_in[3];
  const float* W1    = (const float*)d_in[4];
  const float* b1    = (const float*)d_in[5];
  const float* W2    = (const float*)d_in[6];
  const float* b2    = (const float*)d_in[7];
  const float* Wpost = (const float*)d_in[8];
  const float* bpost = (const float*)d_in[9];
  (void)n_in; (void)out_size; (void)ws_size;

  const int N = in_sizes[0] / 256;   // 100000
  const int E = in_sizes[1] / 2;     // 1600000
  float* out = (float*)d_out;

  // workspace carve-out (256B aligned)
  char* ws = (char*)d_ws;
  auto carve = [&](size_t bytes) {
    char* p = ws;
    ws += (bytes + 255) & ~(size_t)255;
    return p;
  };
  float*          dinv   = (float*)carve((size_t)N * 4);
  float*          bufA   = (float*)carve((size_t)N * 128 * 4);
  float*          bufB   = (float*)carve((size_t)N * 128 * 4);
  float*          logit  = (float*)carve((size_t)N * 40 * 4);
  unsigned short* pWpre  = (unsigned short*)carve(8 * 8 * 512 * 2);
  unsigned short* pW1    = (unsigned short*)carve(8 * 4 * 512 * 2);
  unsigned short* pW2    = (unsigned short*)carve(8 * 4 * 512 * 2);
  unsigned short* pWpost = (unsigned short*)carve(3 * 4 * 512 * 2);

  const int nRT = N / 16;            // 6250, exact

  // 1) degree -> dinv (self-loop folded into init value 1.0)
  gcn_deg_init <<<(N + 255) / 256, 256, 0, stream>>>(dinv, N);
  gcn_deg_count<<<(E + 255) / 256, 256, 0, stream>>>(ei, dinv, E);
  gcn_deg_rsqrt<<<(N + 255) / 256, 256, 0, stream>>>(dinv, N);

  // 2) pack weights into bf16 WMMA B-fragment layout
  gcn_pack_w<<<(8 * 8 * 512 + 255) / 256, 256, 0, stream>>>(Wpre,  pWpre,  256, 128, 8, 8);
  gcn_pack_w<<<(8 * 4 * 512 + 255) / 256, 256, 0, stream>>>(W1,    pW1,    128, 128, 4, 8);
  gcn_pack_w<<<(8 * 4 * 512 + 255) / 256, 256, 0, stream>>>(W2,    pW2,    128, 128, 4, 8);
  gcn_pack_w<<<(3 * 4 * 512 + 255) / 256, 256, 0, stream>>>(Wpost, pWpost, 128,  40, 4, 3);

  const int mmBlocks = (nRT + 3) / 4;

  // 3) h0 = x @ W_pre + b_pre                               -> bufA
  gcn_matmul_wmma<256, 128, 128, true, false>
      <<<mmBlocks, 128, 0, stream>>>(x, pWpre, bpre, bufA, nRT);

  // 4) conv1: t = h0 @ W1                                   -> bufB
  gcn_matmul_wmma<128, 128, 128, false, false>
      <<<mmBlocks, 128, 0, stream>>>(bufA, pW1, nullptr, bufB, nRT);
  //    agg1 = scatter(t) + self-loop + b1                   -> bufA
  gcn_agg_init <<<(N * 32 + 255) / 256, 256, 0, stream>>>(bufB, dinv, b1, bufA, N);
  gcn_agg_edges<<<(E + 7) / 8, 256, 0, stream>>>(bufB, ei, dinv, bufA, E);

  // 5) conv2: t = relu(agg1) @ W2                           -> bufB
  gcn_matmul_wmma<128, 128, 128, false, true>
      <<<mmBlocks, 128, 0, stream>>>(bufA, pW2, nullptr, bufB, nRT);
  //    agg2 = scatter(t) + self-loop + b2                   -> bufA
  gcn_agg_init <<<(N * 32 + 255) / 256, 256, 0, stream>>>(bufB, dinv, b2, bufA, N);
  gcn_agg_edges<<<(E + 7) / 8, 256, 0, stream>>>(bufB, ei, dinv, bufA, E);

  // 6) logits = relu(agg2) @ W_post + b_post                -> logit
  gcn_matmul_wmma<128, 48, 40, true, true>
      <<<mmBlocks, 128, 0, stream>>>(bufA, pWpost, bpost, logit, nRT);

  // 7) log_softmax                                          -> d_out
  gcn_log_softmax<<<(N + 255) / 256, 256, 0, stream>>>(logit, out, N);
}
